// Model_37168646979712
// MI455X (gfx1250) — matrix operand order repaired and checked by reading, not compiled, not run on hardware
//
#include <hip/hip_runtime.h>
#include <hip/hip_bf16.h>

// ---------------------------------------------------------------------------
// Sizes (fixed by the reference)
// ---------------------------------------------------------------------------
#define D_DIM 256
#define VOCAB 32000
#define RESP  1000
#define RESP_PAD 1008           // 63 tiles of 16
#define BATCH 8
#define SEQ   256
#define NBT   (BATCH * SEQ)     // 2048
#define MIJ   (D_DIM * D_DIM)   // 65536

typedef __attribute__((ext_vector_type(16))) __bf16 v16bf;
typedef __attribute__((ext_vector_type(8)))  float  v8f;

union FragBF { v16bf bf; uint4 q[2]; };

__device__ __forceinline__ unsigned short f2bf(float f) {
    unsigned u = __float_as_uint(f);
    unsigned r = u + 0x7FFFu + ((u >> 16) & 1u);   // round-to-nearest-even
    return (unsigned short)(r >> 16);
}

// ---------------------------------------------------------------------------
// Converters / gathers
// ---------------------------------------------------------------------------
__global__ void conv_bf16_kernel(const float* __restrict__ in,
                                 unsigned short* __restrict__ out, int n) {
    for (int i = blockIdx.x * blockDim.x + threadIdx.x; i < n;
         i += gridDim.x * blockDim.x)
        out[i] = f2bf(in[i]);
}

// lin_w [1000,32000] f32 -> bf16 padded to [1008,32000] (pad rows = 0)
__global__ void conv_pad_lw_kernel(const float* __restrict__ in,
                                   unsigned short* __restrict__ out) {
    const int n = RESP_PAD * VOCAB;
    for (int i = blockIdx.x * blockDim.x + threadIdx.x; i < n;
         i += gridDim.x * blockDim.x) {
        int r = i / VOCAB;
        out[i] = (r < RESP) ? f2bf(in[i]) : (unsigned short)0;
    }
}

// v[n][k] = Bm[k][tokens[n]]  (bf16), n = b*SEQ + t
__global__ void gather_v_kernel(const int* __restrict__ tokens,
                                const float* __restrict__ Bm,
                                unsigned short* __restrict__ v) {
    int i = blockIdx.x * blockDim.x + threadIdx.x;   // 524288 threads
    int n = i >> 8, k = i & 255;
    v[i] = f2bf(Bm[(size_t)k * VOCAB + tokens[n]]);
}

// Dm [32000,256] f32 -> DmT [256,32000] bf16 (LDS-tiled transpose)
__global__ __launch_bounds__(256) void transDm_kernel(
        const float* __restrict__ Dm, unsigned short* __restrict__ DmT) {
    __shared__ float tile[32][33];
    const int v0 = blockIdx.x * 32;      // 1000 tiles over VOCAB
    const int d0 = blockIdx.y * 32;      // 8 tiles over D
    const int tx = threadIdx.x & 31, ty = threadIdx.x >> 5;   // 32 x 8
#pragma unroll
    for (int rr = 0; rr < 32; rr += 8)
        tile[ty + rr][tx] = Dm[(size_t)(v0 + ty + rr) * D_DIM + d0 + tx];
    __syncthreads();
#pragma unroll
    for (int rr = 0; rr < 32; rr += 8)
        DmT[(size_t)(d0 + ty + rr) * VOCAB + v0 + tx] = f2bf(tile[tx][ty + rr]);
}

// bias2[r] = sum_v e[v]*lin_w[r,v] + lin_b[r]
__global__ __launch_bounds__(256) void bias2_kernel(
        const float* __restrict__ lin_w, const float* __restrict__ e,
        const float* __restrict__ lin_b, float* __restrict__ bias2) {
    __shared__ float red[256];
    const int r = blockIdx.x;
    float acc = 0.f;
    for (int v = threadIdx.x; v < VOCAB; v += 256)
        acc += e[v] * lin_w[(size_t)r * VOCAB + v];
    red[threadIdx.x] = acc; __syncthreads();
    for (int s = 128; s > 0; s >>= 1) {
        if (threadIdx.x < (unsigned)s) red[threadIdx.x] += red[threadIdx.x + s];
        __syncthreads();
    }
    if (threadIdx.x == 0) bias2[r] = red[0] + lin_b[r];
}

// ---------------------------------------------------------------------------
// GEMM 1: A[n][m] = sum_k v[n][k] * W[m][k]     (M-dim = n, N-dim = m, K = 256)
//   v16bf A-frag: lane row = n, 16 contiguous K (hi lanes +16)
//   v16bf B-frag: lane col = m, 16 contiguous K (hi lanes +16)
// Each wave: one 16-n tile x four 16-m tiles (A-frag reuse), 8 K-steps.
// ---------------------------------------------------------------------------
__global__ __launch_bounds__(256) void gemmA_kernel(
        const unsigned short* __restrict__ Wbf,   // [MIJ][256]
        const unsigned short* __restrict__ vbf,   // [NBT][256]
        unsigned short* __restrict__ Aout) {      // [NBT][MIJ] bf16
    const int wid  = (blockIdx.x << 3) + (threadIdx.x >> 5);   // 131072 waves
    const int lane = threadIdx.x & 31;
    const int hi   = lane >> 4;
    const int l15  = lane & 15;
    const int mg   = wid & 1023;          // 1024 groups of 64 m
    const int nt   = wid >> 10;           // 128 n-tiles
    const int n_base = nt << 4;
    const int m_base = mg << 6;

    v8f c0 = {}, c1 = {}, c2 = {}, c3 = {};
    const size_t vrow = (size_t)(n_base + l15) * D_DIM;

#pragma unroll
    for (int kk = 0; kk < 8; ++kk) {
        const int k0 = kk * 32;
        FragBF a;
        const uint4* pa = (const uint4*)(vbf + vrow + k0 + hi * 16);
        a.q[0] = pa[0];
        a.q[1] = pa[1];
        FragBF b0, b1, b2, b3;
        {
            const uint4* pb = (const uint4*)(Wbf + (size_t)(m_base +  0 + l15) * D_DIM + k0 + hi * 16);
            b0.q[0] = pb[0]; b0.q[1] = pb[1];
        }
        {
            const uint4* pb = (const uint4*)(Wbf + (size_t)(m_base + 16 + l15) * D_DIM + k0 + hi * 16);
            b1.q[0] = pb[0]; b1.q[1] = pb[1];
        }
        {
            const uint4* pb = (const uint4*)(Wbf + (size_t)(m_base + 32 + l15) * D_DIM + k0 + hi * 16);
            b2.q[0] = pb[0]; b2.q[1] = pb[1];
        }
        {
            const uint4* pb = (const uint4*)(Wbf + (size_t)(m_base + 48 + l15) * D_DIM + k0 + hi * 16);
            b3.q[0] = pb[0]; b3.q[1] = pb[1];
        }
        c0 = __builtin_amdgcn_wmma_f32_16x16x32_bf16(false, a.bf, false, b0.bf, (short)0, c0, false, false);
        c1 = __builtin_amdgcn_wmma_f32_16x16x32_bf16(false, a.bf, false, b1.bf, (short)0, c1, false, false);
        c2 = __builtin_amdgcn_wmma_f32_16x16x32_bf16(false, a.bf, false, b2.bf, (short)0, c2, false, false);
        c3 = __builtin_amdgcn_wmma_f32_16x16x32_bf16(false, a.bf, false, b3.bf, (short)0, c3, false, false);
    }
    // D layout: VGPR r -> (Mlocal = r + 8*hi, Nlocal = l15); store bf16 to A[n][m]
#pragma unroll
    for (int r = 0; r < 8; ++r) {
        const size_t nrow = (size_t)(n_base + r + (hi << 3)) * MIJ;
        Aout[nrow + m_base +  0 + l15] = f2bf(c0[r]);
        Aout[nrow + m_base + 16 + l15] = f2bf(c1[r]);
        Aout[nrow + m_base + 32 + l15] = f2bf(c2[r]);
        Aout[nrow + m_base + 48 + l15] = f2bf(c3[r]);
    }
}

// ---------------------------------------------------------------------------
// GEMM 2: F[r][d] = sum_v lin_w[r][v] * Dm[v][d]    (M=r pad 1008, N=d 256, K=32000)
// One 16x16 tile per wave, 1000 K-steps. Output f32.
// ---------------------------------------------------------------------------
__global__ __launch_bounds__(256) void gemmF_kernel(
        const unsigned short* __restrict__ lwbf,   // [1008][32000]
        const unsigned short* __restrict__ DmT,    // [256][32000]
        float* __restrict__ F) {                   // [1008][256] f32
    const int wid  = (blockIdx.x << 3) + (threadIdx.x >> 5);   // 1008 waves
    const int lane = threadIdx.x & 31;
    const int hi   = lane >> 4;
    const int l15  = lane & 15;
    const int rt   = wid >> 4;            // 63 r-tiles
    const int dt   = wid & 15;            // 16 d-tiles
    const int r_base = rt << 4;
    const int d_base = dt << 4;

    v8f c = {};
    const size_t lwrow = (size_t)(r_base + l15) * VOCAB;
    const size_t dtrow = (size_t)(d_base + l15) * VOCAB;

    for (int v0 = 0; v0 < VOCAB; v0 += 32) {
        FragBF a, b;
        const uint4* pa = (const uint4*)(lwbf + lwrow + v0 + hi * 16);
        a.q[0] = pa[0];
        a.q[1] = pa[1];
        const uint4* pb = (const uint4*)(DmT + dtrow + v0 + hi * 16);
        b.q[0] = pb[0];
        b.q[1] = pb[1];
        c = __builtin_amdgcn_wmma_f32_16x16x32_bf16(false, a.bf, false, b.bf, (short)0, c, false, false);
    }
#pragma unroll
    for (int r = 0; r < 8; ++r)
        F[(size_t)(r_base + r + (hi << 3)) * D_DIM + d_base + l15] = c[r];
}

// ---------------------------------------------------------------------------
// Sequential scan: per batch element b (one 1024-thread workgroup),
//   states[b,t] = h;  z_i = sum_j A[b,t][i*256+j] * h_j;  h = tanh(z + c)
// Streams 128 KB/step; prefetches step t+1's row (independent of h).
// ---------------------------------------------------------------------------
__global__ __launch_bounds__(1024) void scan_kernel(
        const unsigned short* __restrict__ A,     // [NBT][MIJ] bf16
        const float* __restrict__ c,
        const float* __restrict__ h0,
        float* __restrict__ states) {             // [NBT][256] f32
    __shared__ float sh_h[256];
    __shared__ float sh_c[256];
    __shared__ float sh_p[1024];
    const int b   = blockIdx.x;
    const int tid = threadIdx.x;
    const int i   = tid >> 2;       // output row 0..255
    const int q   = tid & 3;        // j-quarter
    if (tid < 256) { sh_h[tid] = h0[tid]; sh_c[tid] = c[tid]; }
    __syncthreads();

    for (int t = 0; t < SEQ; ++t) {
        const int n = (b << 8) + t;
        if (tid < 256) states[((size_t)n << 8) + tid] = sh_h[tid];  // pre-update state
        const unsigned short* row = A + ((size_t)n << 16);
        if (t + 1 < SEQ)
            __builtin_prefetch(row + (size_t)(1 << 16) + i * 256 + q * 64, 0, 1);
        const uint4* pr = (const uint4*)(row + i * 256 + q * 64);
        const float* hq = sh_h + q * 64;
        float acc = 0.f;
#pragma unroll
        for (int c8 = 0; c8 < 8; ++c8) {
            uint4 u = pr[c8];
            acc += __uint_as_float(u.x << 16)         * hq[c8 * 8 + 0];
            acc += __uint_as_float(u.x & 0xFFFF0000u) * hq[c8 * 8 + 1];
            acc += __uint_as_float(u.y << 16)         * hq[c8 * 8 + 2];
            acc += __uint_as_float(u.y & 0xFFFF0000u) * hq[c8 * 8 + 3];
            acc += __uint_as_float(u.z << 16)         * hq[c8 * 8 + 4];
            acc += __uint_as_float(u.z & 0xFFFF0000u) * hq[c8 * 8 + 5];
            acc += __uint_as_float(u.w << 16)         * hq[c8 * 8 + 6];
            acc += __uint_as_float(u.w & 0xFFFF0000u) * hq[c8 * 8 + 7];
        }
        sh_p[tid] = acc;
        __syncthreads();
        if (tid < 256) {
            float z = sh_p[tid * 4] + sh_p[tid * 4 + 1] +
                      sh_p[tid * 4 + 2] + sh_p[tid * 4 + 3] + sh_c[tid];
            sh_h[tid] = tanhf(z);
        }
        __syncthreads();
    }
}

// ---------------------------------------------------------------------------
// Fused logits + log_softmax: one block per (b,t).
//   l[r] = states[n] . F[r] + bias2[r];  out = l - logsumexp(l)
// ---------------------------------------------------------------------------
__global__ __launch_bounds__(256) void logits_kernel(
        const float* __restrict__ states, const float* __restrict__ F,
        const float* __restrict__ bias2, float* __restrict__ out) {
    __shared__ float sh_s[256];
    __shared__ float red[256];
    const int n = blockIdx.x;
    const int tid = threadIdx.x;
    sh_s[tid] = states[((size_t)n << 8) + tid];
    __syncthreads();
    float l[4];
#pragma unroll
    for (int j = 0; j < 4; ++j) {
        const int r = tid + j * 256;
        if (r < RESP) {
            const float* Fr = F + (size_t)r * D_DIM;
            float acc = 0.f;
            for (int d = 0; d < D_DIM; d += 4)
                acc += Fr[d] * sh_s[d] + Fr[d + 1] * sh_s[d + 1]
                     + Fr[d + 2] * sh_s[d + 2] + Fr[d + 3] * sh_s[d + 3];
            l[j] = acc + bias2[r];
        } else {
            l[j] = -3.0e38f;
        }
    }
    float m = fmaxf(fmaxf(l[0], l[1]), fmaxf(l[2], l[3]));
    red[tid] = m; __syncthreads();
    for (int s = 128; s > 0; s >>= 1) {
        if (tid < s) red[tid] = fmaxf(red[tid], red[tid + s]);
        __syncthreads();
    }
    const float gmax = red[0]; __syncthreads();
    float se = 0.f;
#pragma unroll
    for (int j = 0; j < 4; ++j) {
        const int r = tid + j * 256;
        if (r < RESP) se += __expf(l[j] - gmax);
    }
    red[tid] = se; __syncthreads();
    for (int s = 128; s > 0; s >>= 1) {
        if (tid < s) red[tid] += red[tid + s];
        __syncthreads();
    }
    const float lse = gmax + __logf(red[0]);
#pragma unroll
    for (int j = 0; j < 4; ++j) {
        const int r = tid + j * 256;
        if (r < RESP) out[(size_t)n * RESP + r] = l[j] - lse;
    }
}

// ---------------------------------------------------------------------------
// Host launcher
// ---------------------------------------------------------------------------
extern "C" void kernel_launch(void* const* d_in, const int* in_sizes, int n_in,
                              void* d_out, int out_size, void* d_ws, size_t ws_size,
                              hipStream_t stream) {
    (void)in_sizes; (void)n_in; (void)out_size; (void)ws_size;
    const int*   tokens = (const int*)  d_in[0];
    const float* W      = (const float*)d_in[1];
    const float* Bm     = (const float*)d_in[2];
    const float* c      = (const float*)d_in[3];
    const float* Dm     = (const float*)d_in[4];
    const float* e      = (const float*)d_in[5];
    const float* h0     = (const float*)d_in[6];
    const float* lin_w  = (const float*)d_in[7];
    const float* lin_b  = (const float*)d_in[8];
    float* out = (float*)d_out;

    char* p = (char*)d_ws;
    auto alloc = [&](size_t bytes) -> char* {
        char* r = p;
        p += (bytes + 255) & ~(size_t)255;
        return r;
    };
    unsigned short* ws_W   = (unsigned short*)alloc((size_t)MIJ * D_DIM * 2);        // 33.5 MB
    unsigned short* ws_v   = (unsigned short*)alloc((size_t)NBT * D_DIM * 2);        //  1  MB
    unsigned short* ws_A   = (unsigned short*)alloc((size_t)NBT * MIJ * 2);          // 268 MB
    unsigned short* ws_DmT = (unsigned short*)alloc((size_t)D_DIM * VOCAB * 2);      // 16  MB
    unsigned short* ws_lw  = (unsigned short*)alloc((size_t)RESP_PAD * VOCAB * 2);   // 64  MB
    float*          ws_F   = (float*)alloc((size_t)RESP_PAD * D_DIM * 4);
    float*          ws_b2  = (float*)alloc((size_t)RESP_PAD * 4);
    float*          ws_st  = (float*)alloc((size_t)NBT * D_DIM * 4);                 //  2  MB

    // Precision converts / gathers (all parallel)
    conv_bf16_kernel   <<<4096, 256, 0, stream>>>(W, ws_W, MIJ * D_DIM);
    gather_v_kernel    <<<NBT, 256, 0, stream>>>(tokens, Bm, ws_v);
    conv_pad_lw_kernel <<<4096, 256, 0, stream>>>(lin_w, ws_lw);
    transDm_kernel     <<<dim3(VOCAB / 32, D_DIM / 32), 256, 0, stream>>>(Dm, ws_DmT);
    bias2_kernel       <<<RESP, 256, 0, stream>>>(lin_w, e, lin_b, ws_b2);

    // Big GEMMs (WMMA bf16 -> f32)
    gemmA_kernel <<<16384, 256, 0, stream>>>(ws_W, ws_v, ws_A);   // 68.7 GFLOP
    gemmF_kernel <<<126,   256, 0, stream>>>(ws_lw, ws_DmT, ws_F); // 16.4 GFLOP

    // Sequential recurrence (8 independent scans), then fused logits+log_softmax
    scan_kernel   <<<BATCH, 1024, 0, stream>>>(ws_A, c, h0, ws_st);
    logits_kernel <<<NBT, 256, 0, stream>>>(ws_st, ws_F, ws_b2, out);
}
